// Proposed_ver1_70815420776606
// MI455X (gfx1250) — matrix operand order repaired and checked by reading, not compiled, not run on hardware
//
#include <hip/hip_runtime.h>
#include <math.h>

// ---------------- problem constants ----------------
#define C_DIM 256      // channel axis (acts as conv batch)
#define N_DIM 32       // sample axis (acts as conv channels)
#define H_DIM 56
#define W_DIM 56
#define HW    (H_DIM * W_DIM)        // 3136
#define HP    58                      // padded (halo) height
#define WP    58                      // padded (halo) width
#define NTILES (HW / 16)              // 196 pixel tiles of 16
#define IMG_ELEMS (HP * WP * N_DIM)   // 107648 bf16 elements
#define IMG_BYTES (IMG_ELEMS * 2)     // 215296 bytes of LDS (needs CDNA5 320KB WGP LDS)
#define SMEM_BYTES (IMG_BYTES + N_DIM * 4)

typedef __bf16 v16bf __attribute__((ext_vector_type(16)));
typedef float  v8f   __attribute__((ext_vector_type(8)));

// =====================================================================
// Kernel 1: router conv (3x3, pad 1) + ReLU + global avg pool, per channel c.
// One workgroup (8 wave32) per c. GEMM: M=32 oc (2 tiles), N=16-pixel tiles,
// K=288 = 9 chunks of 32 (chunk = one (kh,kw) tap across 32 input chans).
// Image staged in LDS as bf16, channel-minor, with zero halo.
// =====================================================================
__global__ __launch_bounds__(256, 1)
void conv_router_kernel(const float* __restrict__ x,
                        const float* __restrict__ conv_w,
                        float* __restrict__ feat) {
  extern __shared__ char smem[];
  __bf16* img     = (__bf16*)smem;
  float*  featsum = (float*)(smem + IMG_BYTES);

  const int c    = blockIdx.x;
  const int tid  = threadIdx.x;
  const int lane = tid & 31;
  const int wave = tid >> 5;

  // Zero whole padded image (halo stays zero), zero per-oc accumulators.
  for (int i = tid; i < IMG_ELEMS; i += 256) img[i] = (__bf16)0.0f;
  if (tid < N_DIM) featsum[tid] = 0.0f;
  __syncthreads();

  // Stage x[:, c, :, :] into LDS as bf16, channel-minor: img[(h+1)*WP + (w+1)][n]
  for (int i = tid; i < N_DIM * HW; i += 256) {
    int n = i / HW;
    int r = i - n * HW;
    int h = r / W_DIM;
    int w = r - h * W_DIM;
    float v = x[((size_t)n * C_DIM + c) * HW + r];
    img[((h + 1) * WP + (w + 1)) * N_DIM + n] = (__bf16)v;
  }
  __syncthreads();

  // Pack weights into registers as WMMA A operands (16x32), contiguous K order:
  //   lane<16 : row=lane,    K in {0..15}
  //   lane>=16: row=lane-16, K in {16..31}
  // K chunk kp = kh*3+kw; within chunk, K_local = input channel n.
  const int row0 = lane & 15;
  const int kb   = (lane >> 4) * 16;
  v16bf A[2][9];
  #pragma unroll
  for (int t = 0; t < 2; ++t) {
    int oc = row0 + 16 * t;
    #pragma unroll
    for (int kp = 0; kp < 9; ++kp) {
      v16bf a;
      #pragma unroll
      for (int i = 0; i < 16; ++i) {
        int n = kb + i;
        a[i] = (__bf16)conv_w[oc * 288 + n * 9 + kp];   // conv_w[oc][n][kh][kw]
      }
      A[t][kp] = a;
    }
  }

  float fa0[8], fa1[8];
  #pragma unroll
  for (int i = 0; i < 8; ++i) { fa0[i] = 0.0f; fa1[i] = 0.0f; }

  // B layout (32x16): lane = column (pixel), element i -> K_local = 16*(lane>>4)+i,
  // which is contiguous in n in our channel-minor LDS image -> one 32B LDS load.
  const int col   = lane & 15;
  const int nbase = (lane >> 4) * 16;

  for (int t = wave; t < NTILES; t += 8) {   // wave-uniform: EXEC all ones for WMMA
    int p = t * 16 + col;
    int h = p / W_DIM;
    int w = p - h * W_DIM;
    v8f acc0 = {0.f, 0.f, 0.f, 0.f, 0.f, 0.f, 0.f, 0.f};
    v8f acc1 = acc0;
    #pragma unroll
    for (int kp = 0; kp < 9; ++kp) {
      int kh = kp / 3;
      int kw = kp - kh * 3;
      const v16bf b = *(const v16bf*)&img[((h + kh) * WP + (w + kw)) * N_DIM + nbase];
      acc0 = __builtin_amdgcn_wmma_f32_16x16x32_bf16(false, A[0][kp], false, b,
                                                     (short)0, acc0, false, false);
      acc1 = __builtin_amdgcn_wmma_f32_16x16x32_bf16(false, A[1][kp], false, b,
                                                     (short)0, acc1, false, false);
    }
    // Fused ReLU + pixel accumulation (C/D layout: elem j -> M = j + 8*(lane>>4))
    #pragma unroll
    for (int i = 0; i < 8; ++i) {
      fa0[i] += fmaxf(acc0[i], 0.0f);
      fa1[i] += fmaxf(acc1[i], 0.0f);
    }
  }

  const int mofs = (lane >> 4) * 8;
  #pragma unroll
  for (int i = 0; i < 8; ++i) {
    atomicAdd(&featsum[i + mofs],      fa0[i]);   // ds_add_f32
    atomicAdd(&featsum[16 + i + mofs], fa1[i]);
  }
  __syncthreads();
  if (tid < N_DIM) feat[c * N_DIM + tid] = featsum[tid] * (1.0f / (float)HW);
}

// =====================================================================
// Kernel 2: fc1 + fc2 + argmax router (one thread per channel), plus
// group counts and zeroing of the global stat accumulators.
// =====================================================================
__global__ __launch_bounds__(256)
void router_fc_kernel(const float* __restrict__ feat,
                      const float* __restrict__ fc1_w, const float* __restrict__ fc1_b,
                      const float* __restrict__ fc2_w, const float* __restrict__ fc2_b,
                      int* __restrict__ sel, float* __restrict__ cntf,
                      float* __restrict__ sums, float* __restrict__ sumsq) {
  __shared__ int ls[C_DIM];
  const int tid = threadIdx.x;          // tid == channel c
  sums[tid]  = 0.0f;                    // zero 32x8 bins (padded region)
  sumsq[tid] = 0.0f;

  float f[32];
  #pragma unroll
  for (int k = 0; k < 32; ++k) f[k] = feat[tid * 32 + k];

  float h[32];
  #pragma unroll
  for (int j = 0; j < 32; ++j) {
    float acc = fc1_b[j];
    #pragma unroll
    for (int k = 0; k < 32; ++k) acc += f[k] * fc1_w[j * 32 + k];
    h[j] = acc;
  }

  float best = -1e30f; int bg = 0;      // softmax is monotone: argmax(logits)
  #pragma unroll
  for (int g = 0; g < 8; ++g) {
    float l = fc2_b[g];
    #pragma unroll
    for (int j = 0; j < 32; ++j) l += h[j] * fc2_w[g * 32 + j];
    if (l > best) { best = l; bg = g; } // strict '>' keeps first max (jnp.argmax)
  }
  sel[tid] = bg;
  ls[tid]  = bg;
  __syncthreads();
  if (tid < 8) {
    int cnt = 0;
    for (int cc = 0; cc < C_DIM; ++cc) cnt += (ls[cc] == tid) ? 1 : 0;
    cntf[tid] = (float)cnt * (float)HW; // elems per sample per group
  }
}

// =====================================================================
// Kernel 3: per-(n,c) block sum & sumsq over HxW, atomically folded into
// per-(n,group) bins. x is hot in the 192MB L2 after kernel 1.
// =====================================================================
__global__ __launch_bounds__(256)
void stats_kernel(const float* __restrict__ x, const int* __restrict__ sel,
                  float* __restrict__ sums, float* __restrict__ sumsq) {
  __shared__ float r1[256];
  __shared__ float r2[256];
  const int b = blockIdx.x;             // 0 .. 32*256-1
  const int n = b >> 8;
  const int c = b & 255;
  const int g = sel[c];
  const float* px = x + ((size_t)n * C_DIM + c) * HW;
  const int tid = threadIdx.x;

  float s1 = 0.0f, s2 = 0.0f;
  for (int i = tid; i < HW; i += 256) {
    float v = px[i];
    s1 += v;
    s2 += v * v;
  }
  r1[tid] = s1; r2[tid] = s2;
  __syncthreads();
  for (int o = 128; o > 0; o >>= 1) {
    if (tid < o) { r1[tid] += r1[tid + o]; r2[tid] += r2[tid + o]; }
    __syncthreads();
  }
  if (tid == 0) {
    atomicAdd(&sums[n * 8 + g],  r1[0]);
    atomicAdd(&sumsq[n * 8 + g], r2[0]);
  }
}

// =====================================================================
// Kernel 4: fold mean / unbiased var / gamma / beta into per-(n,c)
// scale & shift:  out = x*scale + shift.
// =====================================================================
__global__ __launch_bounds__(256)
void coeff_kernel(const int* __restrict__ sel, const float* __restrict__ cntf,
                  const float* __restrict__ sums, const float* __restrict__ sumsq,
                  const float* __restrict__ gamma, const float* __restrict__ beta,
                  float* __restrict__ scale, float* __restrict__ shift) {
  const int idx = blockIdx.x * 256 + threadIdx.x;  // n*256 + c
  const int n = idx >> 8;
  const int c = idx & 255;
  const int g = sel[c];
  float cnt = cntf[g];
  float cs  = fmaxf(cnt, 1.0f);
  float m   = sums[n * 8 + g] / cs;
  float var = (sumsq[n * 8 + g] - cs * m * m) / fmaxf(cs - 1.0f, 1.0f);
  float iv  = 1.0f / sqrtf(var + 1e-5f);
  float sc  = iv * gamma[c];
  scale[idx] = sc;
  shift[idx] = beta[c] - m * sc;
}

// =====================================================================
// Kernel 5: streaming normalize, float4 (HW = 784 float4 per (n,c)).
// =====================================================================
__global__ __launch_bounds__(256)
void norm_kernel(const float* __restrict__ x, const float* __restrict__ scale,
                 const float* __restrict__ shift, float* __restrict__ out) {
  const size_t idx = (size_t)blockIdx.x * 256 + threadIdx.x;  // float4 index
  const int nc = (int)(idx / (HW / 4));
  const float sc = scale[nc];
  const float sh = shift[nc];
  const float4 v = ((const float4*)x)[idx];
  float4 o;
  o.x = v.x * sc + sh;
  o.y = v.y * sc + sh;
  o.z = v.z * sc + sh;
  o.w = v.w * sc + sh;
  ((float4*)out)[idx] = o;
}

// =====================================================================
extern "C" void kernel_launch(void* const* d_in, const int* in_sizes, int n_in,
                              void* d_out, int out_size, void* d_ws, size_t ws_size,
                              hipStream_t stream) {
  const float* x      = (const float*)d_in[0];
  const float* conv_w = (const float*)d_in[1];
  const float* fc1_w  = (const float*)d_in[2];
  const float* fc1_b  = (const float*)d_in[3];
  const float* fc2_w  = (const float*)d_in[4];
  const float* fc2_b  = (const float*)d_in[5];
  const float* gamma  = (const float*)d_in[6];
  const float* beta   = (const float*)d_in[7];
  float* out = (float*)d_out;

  // workspace layout (floats)
  float* ws    = (float*)d_ws;
  float* feat  = ws;                    // 256*32 = 8192
  int*   sel   = (int*)(ws + 8192);     // 256
  float* cntf  = ws + 8448;             // 8
  float* sums  = ws + 8512;             // 256 (32*8 used)
  float* sumsq = ws + 8768;             // 256
  float* scale = ws + 9216;             // 8192
  float* shift = ws + 17408;            // 8192  (total < 110 KB)

  conv_router_kernel<<<C_DIM, 256, SMEM_BYTES, stream>>>(x, conv_w, feat);
  router_fc_kernel<<<1, 256, 0, stream>>>(feat, fc1_w, fc1_b, fc2_w, fc2_b,
                                          sel, cntf, sums, sumsq);
  stats_kernel<<<N_DIM * C_DIM, 256, 0, stream>>>(x, sel, sums, sumsq);
  coeff_kernel<<<N_DIM, 256, 0, stream>>>(sel, cntf, sums, sumsq, gamma, beta,
                                          scale, shift);
  norm_kernel<<<(N_DIM * C_DIM * (HW / 4)) / 256, 256, 0, stream>>>(x, scale, shift, out);
}
